// GradientConflictEngine_87101936763276
// MI455X (gfx1250) — matrix operand order repaired and checked by reading, not compiled, not run on hardware
//
#include <hip/hip_runtime.h>
#include <math.h>

#define E_ 128
#define K_ 65536
#define D_ 1048576
#define KC 128            // bf16 columns staged per slab
#define PITCH_W 66        // LDS row pitch in 32-bit words (64 data words + 2 pad)
#define PITCH_B (PITCH_W * 4u)
#define NSLAB (D_ / KC)   // 8192
#define GEMM_WGS 512
#define ALPHA_THRESH 1e-3f
#define EPS_ 1e-8f

typedef __attribute__((ext_vector_type(16))) __bf16 v16bf;
typedef __attribute__((ext_vector_type(8)))  float  v8f;

// ---------------------------------------------------------------- zero scratch (bf16 dense + f32 gram)
__global__ void zero_ws_kernel(uint4* __restrict__ p, unsigned n16) {
  unsigned i = blockIdx.x * blockDim.x + threadIdx.x;
  const unsigned stride = gridDim.x * blockDim.x;
  uint4 z; z.x = z.y = z.z = z.w = 0u;
  for (; i < n16; i += stride) p[i] = z;
}

// ---------------------------------------------------------------- densify: scatter-add into packed-bf16 dense
// 32-bit CAS RMW on the word containing the target bf16 element
// (lowers to global_atomic_cmpswap_b32; collisions are rare: ~K^2/2D per expert)
__global__ void scatter_pk_kernel(const float* __restrict__ gv,
                                  const long long* __restrict__ gi,
                                  unsigned* __restrict__ denseW) {
  unsigned i = blockIdx.x * blockDim.x + threadIdx.x;
  const unsigned stride = gridDim.x * blockDim.x;
  const unsigned total = (unsigned)E_ * (unsigned)K_;
  for (; i < total; i += stride) {
    const unsigned e = i >> 16;                          // i / K_ (K_ = 2^16)
    const unsigned long long idx  = (unsigned long long)gi[i];
    const unsigned long long elem = ((unsigned long long)e << 20) + idx;
    unsigned* wp = denseW + (elem >> 1);
    const unsigned sh = ((unsigned)elem & 1u) << 4;      // bits [15:0] or [31:16]
    const float v = gv[i];
    unsigned old = *wp, assumed;
    do {
      assumed = old;
      const float cur = __uint_as_float(((assumed >> sh) & 0xFFFFu) << 16);
      unsigned ub = __float_as_uint(cur + v);
      ub += 0x7FFFu + ((ub >> 16) & 1u);                 // RNE to bf16
      const unsigned nw = (assumed & ~(0xFFFFu << sh)) | ((ub >> 16) << sh);
      old = atomicCAS(wp, assumed, nw);
    } while (old != assumed);
  }
}

// ---------------------------------------------------------------- Gram = A·Aᵀ via WMMA bf16, async-to-LDS staging
__global__ void __launch_bounds__(256)
gram_wmma_kernel(const unsigned short* __restrict__ dense, float* __restrict__ gram) {
  __shared__ unsigned tileW[E_ * PITCH_W];     // 128 rows x 128 bf16 (64 words) + pad

  const unsigned tid  = threadIdx.x;
  const unsigned lane = tid & 31u;
  const unsigned wave = tid >> 5;              // 8 waves
  const unsigned half = lane >> 4;             // 0: lanes 0-15, 1: lanes 16-31
  const unsigned lid  = lane & 15u;
  const unsigned rowBase = wave * 16u;         // this wave's C row-stripe

  v8f acc[8] = {};                             // 8 col tiles x 16x16 f32 accum

  const unsigned r  = tid >> 1;                // staging: row 0..127
  const unsigned hs = tid & 1u;                // staging: half-row select (128B each)

  const unsigned ldsBase = (unsigned)(uintptr_t)tileW;
  const unsigned long long gbase = (unsigned long long)(uintptr_t)dense;

  // loop-invariant LDS word indices (chunk/vgpr offsets fold into ds-load immediates)
  // A frag (16x32 bf16): word = idxA + kcW + vv
  const unsigned idxA = (rowBase + lid) * PITCH_W + (half << 3);
  // B frag (32x16 bf16): word = idxB[n] + kcW + vv
  unsigned idxB[8];
  #pragma unroll
  for (unsigned n = 0; n < 8; ++n) idxB[n] = (n * 16u + lid) * PITCH_W + (half << 3);

  for (unsigned s = blockIdx.x; s < NSLAB; s += gridDim.x) {
    const unsigned d0b = s * (KC * 2u);        // byte offset of slab within a row

    // ---- stage slab with CDNA5 async DMA: global bf16 -> LDS, 16B per lane-op
    {
      const unsigned goff  = r * (D_ * 2u) + d0b + hs * 128u;   // global byte offset (GVS)
      const unsigned laddr = ldsBase + r * PITCH_B + hs * 128u; // LDS dest byte addr
      #pragma unroll
      for (unsigned c = 0; c < 8; ++c) {
        asm volatile("global_load_async_to_lds_b128 %0, %1, %2"
                     :: "v"(laddr + c * 16u), "v"(goff + c * 16u), "s"(gbase)
                     : "memory");
      }
    }
    // speculative prefetch of this WG's next slab (global_prefetch_b8)
    if (s + gridDim.x < NSLAB) {
      __builtin_prefetch(dense + (size_t)r * D_ + (size_t)(s + gridDim.x) * KC
                             + (size_t)hs * 64u, 0, 0);
    }
    asm volatile("s_wait_asynccnt 0x0" ::: "memory");
    __syncthreads();

    // ---- 4 k-chunks of 32 bf16 each, B fragments double-buffered to kill
    //      WMMA->VALU WAR hazard nops and overlap ds_loads with WMMA
    #pragma unroll
    for (unsigned chunk = 0; chunk < KC / 32u; ++chunk) {
      const unsigned kcW = chunk * 16u;        // word offset of chunk in row

      union Frag { unsigned w[8]; v16bf v; };
      Frag afr;
      #pragma unroll
      for (unsigned vv = 0; vv < 8; ++vv)
        afr.w[vv] = tileW[idxA + kcW + vv];

      Frag bfr[2];
      #pragma unroll
      for (unsigned vv = 0; vv < 8; ++vv)
        bfr[0].w[vv] = tileW[idxB[0] + kcW + vv];

      #pragma unroll
      for (unsigned n = 0; n < 8; ++n) {
        if (n < 7) {
          #pragma unroll
          for (unsigned vv = 0; vv < 8; ++vv)
            bfr[(n + 1) & 1u].w[vv] = tileW[idxB[n + 1] + kcW + vv];
        }
        acc[n] = __builtin_amdgcn_wmma_f32_16x16x32_bf16(
            false, afr.v, false, bfr[n & 1u].v, (short)0, acc[n], false, false);
      }
    }
    __syncthreads();
  }

  // ---- commit partial Gram (C layout: vgpr rr -> M = rowBase+rr+8*half, N = lid)
  #pragma unroll
  for (unsigned n = 0; n < 8; ++n) {
    #pragma unroll
    for (unsigned rr = 0; rr < 8; ++rr) {
      const unsigned gm = rowBase + rr + half * 8u;
      const unsigned gn = n * 16u + lid;
      atomicAdd(gram + (size_t)gm * E_ + gn, acc[n][rr]);
    }
  }
}

// ---------------------------------------------------------------- cosine conflict reduction
__global__ void reduce_conflict_kernel(const float* __restrict__ gram,
                                       const float* __restrict__ alpha,
                                       float* __restrict__ out) {
  __shared__ float norms[E_];
  __shared__ float red[E_];
  const unsigned i = threadIdx.x;              // 0..127
  norms[i] = sqrtf(gram[i * E_ + i]);
  __syncthreads();
  const float ni = norms[i];
  float local = 0.f;
  for (unsigned j = 0; j < E_; ++j) {
    if (j == i) continue;
    const float a = alpha[i * E_ + j];
    if (a >= ALPHA_THRESH) {
      const float cosv = gram[i * E_ + j] / (ni * norms[j] + EPS_);
      local += a * fmaxf(-cosv, 0.f);
    }
  }
  red[i] = local;
  __syncthreads();
  for (unsigned off = 64; off > 0; off >>= 1) {
    if (i < off) red[i] += red[i + off];
    __syncthreads();
  }
  if (i == 0) out[0] = red[0];
}

// ---------------------------------------------------------------- launch
extern "C" void kernel_launch(void* const* d_in, const int* in_sizes, int n_in,
                              void* d_out, int out_size, void* d_ws, size_t ws_size,
                              hipStream_t stream) {
  (void)in_sizes; (void)n_in; (void)out_size; (void)ws_size;

  const float*     alpha = (const float*)d_in[0];       // [E,E] f32
  const float*     gv    = (const float*)d_in[1];       // [E,K] f32
  const long long* gi    = (const long long*)d_in[2];   // [E,K] i64

  unsigned short* dense = (unsigned short*)d_ws;        // [E,D] bf16 (256 MB)
  float* gram = (float*)((char*)d_ws + (size_t)E_ * (size_t)D_ * 2u);  // [E,E] f32
  float* out  = (float*)d_out;                          // scalar f32

  const unsigned n16 = (unsigned)(((size_t)E_ * D_ * 2u + (size_t)E_ * E_ * 4u) / 16u);
  zero_ws_kernel<<<4096, 256, 0, stream>>>((uint4*)d_ws, n16);
  scatter_pk_kernel<<<8192, 256, 0, stream>>>(gv, gi, (unsigned*)dense);
  gram_wmma_kernel<<<GEMM_WGS, 256, 0, stream>>>(dense, gram);
  reduce_conflict_kernel<<<1, E_, 0, stream>>>(gram, alpha, out);
}
